// VGRNN_76914274337176
// MI455X (gfx1250) — matrix operand order repaired and checked by reading, not compiled, not run on hardware
//
#include <hip/hip_runtime.h>
#include <math.h>

typedef _Float16 h16;
typedef __attribute__((ext_vector_type(16))) _Float16 v16h;
typedef __attribute__((ext_vector_type(8)))  float    v8f;

static constexpr int TPB = 256;

// ---------------- elementwise / utility kernels ----------------

__global__ void k_zero(float* p, long long n) {
  long long st = (long long)gridDim.x * blockDim.x;
  for (long long i = (long long)blockIdx.x * blockDim.x + threadIdx.x; i < n; i += st)
    p[i] = 0.0f;
}

__global__ void k_count_deg(float* deg, const int* __restrict__ row, int e) {
  int st = gridDim.x * blockDim.x;
  for (int i = blockIdx.x * blockDim.x + threadIdx.x; i < e; i += st)
    atomicAdd(&deg[row[i]], 1.0f);
}

__global__ void k_dinv(float* d, int n) {
  int st = gridDim.x * blockDim.x;
  for (int i = blockIdx.x * blockDim.x + threadIdx.x; i < n; i += st)
    d[i] = rsqrtf(d[i] + 1.0f);   // +1 self loop => always > 0
}

__global__ void k_concat2(float* __restrict__ dst, const float* __restrict__ a,
                          const float* __restrict__ b, int n, int ma, int mb) {
  int m = ma + mb;
  long long tot = (long long)n * m;
  long long st = (long long)gridDim.x * blockDim.x;
  for (long long i = (long long)blockIdx.x * blockDim.x + threadIdx.x; i < tot; i += st) {
    int r = (int)(i / m), c = (int)(i % m);
    dst[i] = (c < ma) ? a[(long long)r * ma + c] : b[(long long)r * mb + (c - ma)];
  }
}

// Pack a K x M f32 matrix (optionally stored transposed, i.e. M x K row-major)
// into the exact WMMA 16x16x32 f16 B-register layout:
//   dst[ ((jb*(K/32)+kc)*32 + g*16 + n)*16 + j ]  holds  B[kc*32 + (j>>3)*16 + g*8 + (j&7)][jb*16 + n]
// so each lane's B fragment is one contiguous 32B run.
__global__ void k_pack_b(h16* __restrict__ dst, const float* __restrict__ src,
                         int K, int M, int trans) {
  int kcn = K >> 5;
  long long tot = (long long)K * M;
  long long st = (long long)gridDim.x * blockDim.x;
  for (long long i = (long long)blockIdx.x * blockDim.x + threadIdx.x; i < tot; i += st) {
    int k = (int)(i / M), m = (int)(i % M);
    float v = trans ? src[(long long)m * K + k] : src[(long long)k * M + m];
    int jb = m >> 4, n = m & 15;
    int kc = k >> 5, g = (k >> 3) & 1, j = (k & 7) | (((k >> 4) & 1) << 3);
    dst[((((long long)jb * kcn + kc) * 32) + g * 16 + n) * 16 + j] = (h16)v;
  }
}

// ---------------- WMMA GEMM: C[4096 x M] = A[4096 x K](f32) @ Bpacked(f16) ----------------
// 2x2 register blocking: each wave computes a 32x32 output block (four 16x16 WMMA
// tiles). __launch_bounds__(128,1) relaxes the VGPR budget so all 12 b128 loads
// per K-chunk can hold independent destination registers and clause/overlap.
// A per ISA A-layout (lane m = L&15, g = L>>4: halves 0..7 -> K=g*8+j,
// halves 8..15 -> K=16+g*8+j), converted f32->f16 in-register.
__device__ __forceinline__ v16h cvt_frag(float4 p0, float4 p1, float4 p2, float4 p3) {
  v16h Ah;
  Ah[0]  = (h16)p0.x; Ah[1]  = (h16)p0.y; Ah[2]  = (h16)p0.z; Ah[3]  = (h16)p0.w;
  Ah[4]  = (h16)p1.x; Ah[5]  = (h16)p1.y; Ah[6]  = (h16)p1.z; Ah[7]  = (h16)p1.w;
  Ah[8]  = (h16)p2.x; Ah[9]  = (h16)p2.y; Ah[10] = (h16)p2.z; Ah[11] = (h16)p2.w;
  Ah[12] = (h16)p3.x; Ah[13] = (h16)p3.y; Ah[14] = (h16)p3.z; Ah[15] = (h16)p3.w;
  return Ah;
}

__global__ void __launch_bounds__(128, 1)
k_gemm_wmma(const float* __restrict__ A, const h16* __restrict__ Bp,
            float* __restrict__ C, int K, int M,
            const float* __restrict__ bias, int relu) {
  const int lane = threadIdx.x & 31;
  const int wave = threadIdx.x >> 5;
  const int n16  = lane & 15;
  const int g    = lane >> 4;
  const int ti   = blockIdx.y;                 // 32-row block
  const int tw   = (blockIdx.x << 2) + wave;   // 32-col block
  const int kcn  = K >> 5;
  const int tj0  = tw * 2, tj1 = tj0 + 1;

  const float* arow0 = A + (long long)(ti * 32 + n16) * K;
  const float* arow1 = A + (long long)(ti * 32 + 16 + n16) * K;
  const h16*   bp0   = Bp + (((long long)tj0 * kcn) * 32 + lane) * 16;
  const h16*   bp1   = Bp + (((long long)tj1 * kcn) * 32 + lane) * 16;

  v8f acc00 = {}, acc01 = {}, acc10 = {}, acc11 = {};
  for (int kc = 0; kc < kcn; ++kc) {
    const int ko = (kc << 5) + (g << 3);
    if (kc + 1 < kcn) {
      __builtin_prefetch(arow0 + ko + 32, 0, 1);
      __builtin_prefetch(arow1 + ko + 32, 0, 1);
    }
    // issue all global loads first (independent destinations)
    const float4* pa0 = (const float4*)(arow0 + ko);   // frag0: [0],[1] then +16f = [4],[5]
    const float4* pa1 = (const float4*)(arow1 + ko);
    float4 q0 = pa0[0], q1 = pa0[1], q2 = pa0[4], q3 = pa0[5];
    float4 r0 = pa1[0], r1 = pa1[1], r2 = pa1[4], r3 = pa1[5];
    v16h B0  = *(const v16h*)(bp0 + (long long)kc * 512);
    v16h B1  = *(const v16h*)(bp1 + (long long)kc * 512);
    // then convert
    v16h Ah0 = cvt_frag(q0, q1, q2, q3);
    v16h Ah1 = cvt_frag(r0, r1, r2, r3);
    // then 4 WMMAs sharing the fragments
    acc00 = __builtin_amdgcn_wmma_f32_16x16x32_f16(false, Ah0, false, B0, (short)0, acc00, false, false);
    acc01 = __builtin_amdgcn_wmma_f32_16x16x32_f16(false, Ah0, false, B1, (short)0, acc01, false, false);
    acc10 = __builtin_amdgcn_wmma_f32_16x16x32_f16(false, Ah1, false, B0, (short)0, acc10, false, false);
    acc11 = __builtin_amdgcn_wmma_f32_16x16x32_f16(false, Ah1, false, B1, (short)0, acc11, false, false);
  }

  const int c0 = tj0 * 16 + n16;
  const int c1 = tj1 * 16 + n16;
  const float bv0 = bias ? bias[c0] : 0.0f;
  const float bv1 = bias ? bias[c1] : 0.0f;
#pragma unroll
  for (int r = 0; r < 8; ++r) {
    const long long row0 = (long long)(ti * 32 + r + (g << 3)) * M;       // C lane: col=L&15, row=r+8g
    const long long row1 = (long long)(ti * 32 + 16 + r + (g << 3)) * M;
    float v00 = acc00[r] + bv0, v01 = acc01[r] + bv1;
    float v10 = acc10[r] + bv0, v11 = acc11[r] + bv1;
    if (relu) {
      v00 = fmaxf(v00, 0.0f); v01 = fmaxf(v01, 0.0f);
      v10 = fmaxf(v10, 0.0f); v11 = fmaxf(v11, 0.0f);
    }
    C[row0 + c0] = v00; C[row0 + c1] = v01;
    C[row1 + c0] = v10; C[row1 + c1] = v11;
  }
}

// ---------------- GCN scatter: out[row[e]] += dinv[row]*dinv[col] * y[col[e]] ----------------
__global__ void k_scatter(float* __restrict__ out, const float* __restrict__ y,
                          const int* __restrict__ row, const int* __restrict__ col,
                          const float* __restrict__ dinv, int E, int M) {
  const int m4 = M >> 2;
  const long long tot = (long long)E * m4;
  const long long st  = (long long)gridDim.x * blockDim.x;
  for (long long i = (long long)blockIdx.x * blockDim.x + threadIdx.x; i < tot; i += st) {
    int e = (int)(i / m4);
    int c = ((int)(i % m4)) << 2;
    int r = row[e], cc = col[e];
    float nrm = dinv[r] * dinv[cc];
    float4 v = *(const float4*)(y + (long long)cc * M + c);
    float* o = out + (long long)r * M + c;
    atomicAdd(o + 0, nrm * v.x);
    atomicAdd(o + 1, nrm * v.y);
    atomicAdd(o + 2, nrm * v.z);
    atomicAdd(o + 3, nrm * v.w);
  }
}

__global__ void k_selfloop(float* __restrict__ out, const float* __restrict__ y,
                           const float* __restrict__ dinv, int N, int M) {
  long long tot = (long long)N * M;
  long long st  = (long long)gridDim.x * blockDim.x;
  for (long long i = (long long)blockIdx.x * blockDim.x + threadIdx.x; i < tot; i += st) {
    int r = (int)(i / M);
    float d = dinv[r];
    out[i] += d * d * y[i];
  }
}

__global__ void k_relu(float* p, long long n) {
  long long st = (long long)gridDim.x * blockDim.x;
  for (long long i = (long long)blockIdx.x * blockDim.x + threadIdx.x; i < n; i += st)
    p[i] = fmaxf(p[i], 0.0f);
}

__global__ void k_softplus(float* p, long long n) {
  long long st = (long long)gridDim.x * blockDim.x;
  for (long long i = (long long)blockIdx.x * blockDim.x + threadIdx.x; i < n; i += st) {
    float x = p[i];
    p[i] = (x > 20.0f) ? x : log1pf(__expf(x));
  }
}

__global__ void k_z(float* __restrict__ z, float* __restrict__ zd,
                    const float* __restrict__ eps, const float* __restrict__ sd,
                    const float* __restrict__ mu, const float* __restrict__ du,
                    long long n) {
  long long st = (long long)gridDim.x * blockDim.x;
  for (long long i = (long long)blockIdx.x * blockDim.x + threadIdx.x; i < n; i += st) {
    float zz = eps[i] * sd[i] + mu[i];
    z[i] = zz;
    zd[i] = (du[i] < 0.5f) ? 2.0f * zz : 0.0f;
  }
}

__device__ __forceinline__ float sigmoidf_(float x) { return 1.0f / (1.0f + __expf(-x)); }

__global__ void k_lstm(float* __restrict__ h, float* __restrict__ c,
                       const float* __restrict__ gi, const float* __restrict__ gf,
                       const float* __restrict__ go, const float* __restrict__ gc,
                       long long n) {
  long long st = (long long)gridDim.x * blockDim.x;
  for (long long i = (long long)blockIdx.x * blockDim.x + threadIdx.x; i < n; i += st) {
    float I  = sigmoidf_(gi[i]);
    float F  = sigmoidf_(gf[i]);
    float O  = sigmoidf_(go[i]);
    float Ct = tanhf(gc[i]);
    float hold = h[i], cold = c[i];
    c[i] = F * hold + I * Ct;     // faithful: F * old h
    h[i] = O * tanhf(cold);       // faithful: old c
  }
}

// ---------------- host orchestration ----------------

extern "C" void kernel_launch(void* const* d_in, const int* in_sizes, int n_in,
                              void* d_out, int out_size, void* d_ws, size_t ws_size,
                              hipStream_t stream) {
  (void)in_sizes; (void)n_in; (void)out_size; (void)ws_size;
  constexpr int T = 6, N = 4096, XDi = 64, H = 256, Z = 128, E = 131072;

  const float* x         = (const float*)d_in[0];
  const int*   edges     = (const int*)d_in[1];
  const float* eps       = (const float*)d_in[2];
  const float* du        = (const float*)d_in[3];
  const float* phi_x_w   = (const float*)d_in[4];
  const float* phi_x_b   = (const float*)d_in[5];
  const float* phi_z_w   = (const float*)d_in[6];
  const float* phi_z_b   = (const float*)d_in[7];
  const float* enc_w     = (const float*)d_in[8];
  const float* enc_mu_w  = (const float*)d_in[9];
  const float* enc_lv_w  = (const float*)d_in[10];
  const float* prior_w   = (const float*)d_in[11];
  const float* prior_b   = (const float*)d_in[12];
  const float* prior_mu_w= (const float*)d_in[13];
  const float* prior_mu_b= (const float*)d_in[14];
  // d_in[15], d_in[16]: prior_lv_* -> result unused in returns, skipped
  const float* gru_wx    = (const float*)d_in[17];
  const float* gru_wh    = (const float*)d_in[18];

  float* out_enc_mu   = (float*)d_out;                          // T*N*Z
  float* out_prior_mu = out_enc_mu + (long long)T * N * Z;      // T*N*Z
  float* out_h        = out_prior_mu + (long long)T * N * Z;    // N*H
  float* out_dec      = out_h + (long long)N * H;               // N*N

  char* wsp = (char*)d_ws;
  auto carve = [&](size_t bytes) -> char* {
    char* p = wsp; wsp += (bytes + 255) & ~(size_t)255; return p;
  };
  h16* phi_x_p   = (h16*)carve((size_t)XDi * H * 2);
  h16* phi_z_p   = (h16*)carve((size_t)Z * H * 2);
  h16* enc_p     = (h16*)carve((size_t)2 * H * H * 2);
  h16* enc_mu_p  = (h16*)carve((size_t)H * Z * 2);
  h16* enc_lv_p  = (h16*)carve((size_t)H * Z * 2);
  h16* prior_p   = (h16*)carve((size_t)H * H * 2);
  h16* prior_mu_p= (h16*)carve((size_t)H * Z * 2);
  h16* wx_p      = (h16*)carve((size_t)4 * 2 * H * H * 2);
  h16* wh_p      = (h16*)carve((size_t)4 * H * H * 2);
  h16* zd_p      = (h16*)carve((size_t)Z * N * 2);
  float* dinv    = (float*)carve((size_t)N * 4);
  float* hbuf    = (float*)carve((size_t)N * H * 4);
  float* cbuf    = (float*)carve((size_t)N * H * 4);
  float* phix    = (float*)carve((size_t)N * H * 4);
  float* xin     = (float*)carve((size_t)N * 2 * H * 4);
  float* tmp     = (float*)carve((size_t)N * H * 4);
  float* enct    = (float*)carve((size_t)N * H * 4);
  float* encsd   = (float*)carve((size_t)N * Z * 4);
  float* priort  = (float*)carve((size_t)N * H * 4);
  float* zbuf    = (float*)carve((size_t)N * Z * 4);
  float* zdrop   = (float*)carve((size_t)N * Z * 4);
  float* phiz    = (float*)carve((size_t)N * H * 4);
  float* gate[4];
  for (int i = 0; i < 4; ++i) gate[i] = (float*)carve((size_t)N * H * 4);

  auto blocks = [](long long n, int tpb) -> int {
    long long b = (n + tpb - 1) / tpb;
    if (b > 65536) b = 65536;
    if (b < 1) b = 1;
    return (int)b;
  };
  auto zero = [&](float* p, long long n) {
    k_zero<<<blocks(n, TPB), TPB, 0, stream>>>(p, n);
  };
  auto pack = [&](h16* dst, const float* src, int K, int M, int trans) {
    k_pack_b<<<blocks((long long)K * M, TPB), TPB, 0, stream>>>(dst, src, K, M, trans);
  };
  auto gemm = [&](const float* A, const h16* Bp, float* C, int K, int M,
                  const float* bias, int relu) {
    dim3 grid(M / 128, N / 32);   // 2x2 tiles per wave, 4 waves per block
    k_gemm_wmma<<<grid, dim3(128), 0, stream>>>(A, Bp, C, K, M, bias, relu);
  };

  // ---- pack all weights into WMMA B layout (f16) ----
  pack(phi_x_p,   phi_x_w,    XDi,   H, 0);
  pack(phi_z_p,   phi_z_w,    Z,     H, 0);
  pack(enc_p,     enc_w,      2 * H, H, 0);
  pack(enc_mu_p,  enc_mu_w,   H,     Z, 0);
  pack(enc_lv_p,  enc_lv_w,   H,     Z, 0);
  pack(prior_p,   prior_w,    H,     H, 0);
  pack(prior_mu_p, prior_mu_w, H,    Z, 0);
  for (int g = 0; g < 4; ++g) {
    pack(wx_p + (size_t)g * 2 * H * H, gru_wx + (size_t)g * 2 * H * H, 2 * H, H, 0);
    pack(wh_p + (size_t)g * H * H,     gru_wh + (size_t)g * H * H,     H,     H, 0);
  }

  zero(hbuf, (long long)N * H);
  zero(cbuf, (long long)N * H);

  for (int t = 0; t < T; ++t) {
    const int* rowp = edges + (long long)t * 2 * E;
    const int* colp = rowp + E;

    // degree -> dinv (in place), self loop folded in as +1
    zero(dinv, N);
    k_count_deg<<<blocks(E, TPB), TPB, 0, stream>>>(dinv, rowp, E);
    k_dinv<<<blocks(N, TPB), TPB, 0, stream>>>(dinv, N);

    auto gcn = [&](const float* Aact, const h16* Wp, int K, int M, float* outb, bool zo) {
      if (zo) zero(outb, (long long)N * M);
      gemm(Aact, Wp, tmp, K, M, nullptr, 0);
      k_scatter<<<blocks((long long)E * (M >> 2), TPB), TPB, 0, stream>>>(
          outb, tmp, rowp, colp, dinv, E, M);
      k_selfloop<<<blocks((long long)N * M, TPB), TPB, 0, stream>>>(outb, tmp, dinv, N, M);
    };

    // phi_x_t = relu(x[t] @ phi_x_w + b)
    gemm(x + (long long)t * N * XDi, phi_x_p, phix, XDi, H, phi_x_b, 1);

    // enc_t = relu(gcn(concat(phi_x, h), enc_w))
    k_concat2<<<blocks((long long)N * 2 * H, TPB), TPB, 0, stream>>>(xin, phix, hbuf, N, H, H);
    gcn(xin, enc_p, 2 * H, H, enct, true);
    k_relu<<<blocks((long long)N * H, TPB), TPB, 0, stream>>>(enct, (long long)N * H);

    // enc_mu (direct to output), enc_std = softplus(gcn(enc_t, enc_lv_w))
    float* mu_t = out_enc_mu + (long long)t * N * Z;
    gcn(enct, enc_mu_p, H, Z, mu_t, true);
    gcn(enct, enc_lv_p, H, Z, encsd, true);
    k_softplus<<<blocks((long long)N * Z, TPB), TPB, 0, stream>>>(encsd, (long long)N * Z);

    // prior path (prior_lv unused)
    gemm(hbuf, prior_p, priort, H, H, prior_b, 1);
    gemm(priort, prior_mu_p, out_prior_mu + (long long)t * N * Z, H, Z, prior_mu_b, 0);

    // z_t = eps*std + mu ; z_drop
    k_z<<<blocks((long long)N * Z, TPB), TPB, 0, stream>>>(
        zbuf, zdrop, eps + (long long)t * N * Z, encsd, mu_t,
        du + (long long)t * N * Z, (long long)N * Z);

    // phi_z_t = relu(z @ phi_z_w + b)
    gemm(zbuf, phi_z_p, phiz, Z, H, phi_z_b, 1);

    // graph-LSTM gates: gate = gcn(concat(phi_x, phi_z), wx[g]) + gcn(h, wh[g])
    k_concat2<<<blocks((long long)N * 2 * H, TPB), TPB, 0, stream>>>(xin, phix, phiz, N, H, H);
    for (int g = 0; g < 4; ++g) {
      gcn(xin,  wx_p + (size_t)g * 2 * H * H, 2 * H, H, gate[g], true);
      gcn(hbuf, wh_p + (size_t)g * H * H,     H,     H, gate[g], false);  // accumulate
    }
    k_lstm<<<blocks((long long)N * H, TPB), TPB, 0, stream>>>(
        hbuf, cbuf, gate[0], gate[1], gate[2], gate[3], (long long)N * H);

    // decoder only needed at last step: dec = zdrop @ zdrop^T via same WMMA kernel
    if (t == T - 1) {
      pack(zd_p, zdrop, Z, N, 1);               // B[k][j] = zdrop[j][k]
      gemm(zdrop, zd_p, out_dec, Z, N, nullptr, 0);
    }
  }

  hipMemcpyAsync(out_h, hbuf, (size_t)N * H * sizeof(float),
                 hipMemcpyDeviceToDevice, stream);
}
